// QENet_4544075399430
// MI455X (gfx1250) — matrix operand order repaired and checked by reading, not compiled, not run on hardware
//
#include <hip/hip_runtime.h>
#include <math.h>

typedef __attribute__((ext_vector_type(16))) _Float16 v16h;
typedef __attribute__((ext_vector_type(8)))  _Float16 v8h;
typedef __attribute__((ext_vector_type(8)))  float    v8f;

#define B_   4
#define N_   256
#define K_   32
#define NC_  8
#define M_   16
#define D1_  256
#define D2_  512
#define ROWS 256          // K_*NC_ rows per (b,n)
#define CH   32           // rows per chunk in MLP kernel
#define NCHUNK 8

// ---------------- small device helpers ----------------

__device__ __forceinline__ float hash_unit(unsigned int x) {
  x ^= x >> 17; x *= 0xed5ad4bbu; x ^= x >> 11; x *= 0xac4c1b51u;
  x ^= x >> 15; x *= 0x31848babu; x ^= x >> 14;
  return (float)x * (2.0f / 4294967296.0f) - 1.0f;
}

__device__ __forceinline__ void normalize4(float u[4]) {
  float n = sqrtf(u[0]*u[0] + u[1]*u[1] + u[2]*u[2] + u[3]*u[3]);
  float inv = 1.0f / fmaxf(n, 1e-12f);
  u[0] *= inv; u[1] *= inv; u[2] *= inv; u[3] *= inv;
}

// symmetric 4x4, packed upper triangle: c0..c9
__device__ __forceinline__ void sym_matvec(const float c[10], const float u[4], float o[4]) {
  o[0] = c[0]*u[0] + c[1]*u[1] + c[2]*u[2] + c[3]*u[3];
  o[1] = c[1]*u[0] + c[4]*u[1] + c[5]*u[2] + c[6]*u[3];
  o[2] = c[2]*u[0] + c[5]*u[1] + c[7]*u[2] + c[8]*u[3];
  o[3] = c[3]*u[0] + c[6]*u[1] + c[8]*u[2] + c[9]*u[3];
}

// quat_avg power iteration: v2 = N(C N(C N(C u0))) since cov is symmetric
__device__ __forceinline__ void power_iter(const float c[10], unsigned int seed, float caps[4]) {
  float u[4];
  #pragma unroll
  for (int i = 0; i < 4; ++i) u[i] = hash_unit(seed + 0x9E3779B9u * (unsigned)(i + 1));
  normalize4(u);
  #pragma unroll
  for (int it = 0; it < 3; ++it) {
    float w[4];
    sym_matvec(c, u, w);
    normalize4(w);
    u[0] = w[0]; u[1] = w[1]; u[2] = w[2]; u[3] = w[3];
  }
  caps[0] = u[0]; caps[1] = u[1]; caps[2] = u[2]; caps[3] = u[3];
}

__device__ __forceinline__ void cov_accum(float c[10], float w, const float q[4]) {
  float w0 = w*q[0], w1 = w*q[1], w2 = w*q[2], w3 = w*q[3];
  c[0] += w0*q[0]; c[1] += w0*q[1]; c[2] += w0*q[2]; c[3] += w0*q[3];
  c[4] += w1*q[1]; c[5] += w1*q[2]; c[6] += w1*q[3];
  c[7] += w2*q[2]; c[8] += w2*q[3]; c[9] += w3*q[3];
}

__device__ __forceinline__ void cov_reduce32(float c[10]) {
  #pragma unroll
  for (int k = 0; k < 10; ++k) {
    float x = c[k];
    x += __shfl_xor(x, 1, 32);
    x += __shfl_xor(x, 2, 32);
    x += __shfl_xor(x, 4, 32);
    x += __shfl_xor(x, 8, 32);
    x += __shfl_xor(x, 16, 32);
    c[k] = x;
  }
}

__device__ __forceinline__ float quat_dist(const float a[4], const float b[4]) {
  float d = a[0]*b[0] + a[1]*b[1] + a[2]*b[2] + a[3]*b[3];
  d = fminf(fmaxf(d, -1.0f + 1e-7f), 1.0f - 1e-7f);
  return 2.0f * acosf(d);
}

// ---------------- prep kernels ----------------

// transpose + f32->f16 for W2 (256x512 -> 512x256) and W3 (512x64 -> 64x512)
__global__ void qenet_prep_weights(const float* __restrict__ W2, const float* __restrict__ W3,
                                   _Float16* __restrict__ w2t, _Float16* __restrict__ w3t) {
  int i = blockIdx.x * 256 + threadIdx.x;
  if (i < D2_ * D1_) {
    int n = i / D1_, k = i % D1_;
    w2t[i] = (_Float16)W2[k * D2_ + n];
  } else {
    int j = i - D2_ * D1_;
    if (j < 64 * D2_) {
      int n = j / D2_, k = j % D2_;
      w3t[j] = (_Float16)W3[k * 64 + n];
    }
  }
}

// miu: per-(b,n,nc) average quaternion over K capsules (unweighted cov + power iter)
__global__ void qenet_miu(const float* __restrict__ caps_in, float* __restrict__ miu) {
  int t = blockIdx.x * 256 + threadIdx.x;
  if (t >= B_ * N_ * NC_) return;
  int bn = t >> 3, nc = t & 7;
  float c[10];
  #pragma unroll
  for (int k = 0; k < 10; ++k) c[k] = 0.0f;
  for (int k = 0; k < K_; ++k) {
    const float* qp = caps_in + (((size_t)bn * K_ + k) * NC_ + nc) * 4;
    float q[4] = { qp[0], qp[1], qp[2], qp[3] };
    cov_accum(c, 1.0f, q);
  }
  float u[4];
  power_iter(c, 0xA511E9B3u ^ ((unsigned)t * 0x2E1B2143u), u);
  float* mp = miu + (size_t)t * 4;
  mp[0] = u[0]; mp[1] = u[1]; mp[2] = u[2]; mp[3] = u[3];
}

// x3 = vec( conj(miu) * (0,x) * miu ) per (b,n,k,nc)
__global__ void qenet_rotate(const float* __restrict__ input_x, const float* __restrict__ miu,
                             float* __restrict__ x3) {
  int t = blockIdx.x * 256 + threadIdx.x;
  if (t >= B_ * N_ * K_ * NC_) return;
  int bn = t >> 8, kn = t & 255;
  int k = kn >> 3, nc = kn & 7;
  const float* mp = miu + ((size_t)bn * NC_ + nc) * 4;
  float m0 = mp[0], m1 = mp[1], m2 = mp[2], m3 = mp[3];
  const float* xp = input_x + ((size_t)bn * K_ + k) * 3;
  float p1 = xp[0], p2 = xp[1], p3 = xp[2];
  float a0 = m0, a1 = -m1, a2 = -m2, a3 = -m3;
  // r = conj(m) * (0,p)
  float r0 = -a1*p1 - a2*p2 - a3*p3;
  float r1 =  a0*p1 + a2*p3 - a3*p2;
  float r2 =  a0*p2 + a3*p1 - a1*p3;
  float r3 =  a0*p3 + a1*p2 - a2*p1;
  // o = r * m (keep vector part)
  float o1 = r0*m1 + m0*r1 + r2*m3 - r3*m2;
  float o2 = r0*m2 + m0*r2 + r3*m1 - r1*m3;
  float o3 = r0*m3 + m0*r3 + r1*m2 - r2*m1;
  float* op = x3 + (size_t)t * 3;
  op[0] = o1; op[1] = o2; op[2] = o3;
}

// ---------------- fused WMMA MLP + votes + routing ----------------
// one workgroup (8 waves) per (b,n); 8 chunks of 32 rows through the MLP,
// votes kept in LDS (64 KB), then per-wave dynamic routing, no global bounce.
__global__ __launch_bounds__(256) void qenet_mlp_route(
    const float* __restrict__ x3, const float* __restrict__ W1, const float* __restrict__ b1,
    const _Float16* __restrict__ w2t, const float* __restrict__ b2,
    const _Float16* __restrict__ w3t, const float* __restrict__ b3,
    const float* __restrict__ caps_in, const float* __restrict__ input_a,
    float* __restrict__ out) {
  __shared__ _Float16 h1[CH * D1_];          // 16 KB
  __shared__ _Float16 h2[CH * D2_];          // 32 KB
  __shared__ float    tl[CH * 64];           //  8 KB
  __shared__ float    votes_l[M_ * ROWS * 4];//  64 KB
  __shared__ float    a_l[ROWS];             //  1 KB   -> 121 KB total (<=320 KB WGP LDS)
  const int bn   = blockIdx.x;
  const int tid  = threadIdx.x;
  const int wave = tid >> 5;
  const int lane = tid & 31;
  const int lrow = lane & 15;     // M (for A) or N (for B/C) within tile
  const int lhi  = lane >> 4;     // selects K-half per WMMA register layout

  a_l[tid] = input_a[(size_t)bn * ROWS + tid];

  for (int chunk = 0; chunk < NCHUNK; ++chunk) {
    // ---- layer 1: (32x3) @ (3x256) + b1, VALU ----
    {
      int r = tid & 31;
      int cbase = (tid >> 5) * 32;
      const float* xp = x3 + ((size_t)bn * ROWS + chunk * CH + r) * 3;
      float x0 = xp[0], x1 = xp[1], x2 = xp[2];
      #pragma unroll
      for (int c = 0; c < 32; ++c) {
        int col = cbase + c;
        float h = fmaf(x0, W1[col], fmaf(x1, W1[D1_ + col], fmaf(x2, W1[2 * D1_ + col], b1[col])));
        h1[r * D1_ + col] = (_Float16)h;
      }
    }
    __syncthreads();

    // ---- layer 2: (32x256) @ (256x512), v_wmma_f32_16x16x32_f16 ----
    // B fragments are chunk-invariant: compiler keeps the wave's W2 tile
    // resident in high VGPRs across the whole kernel.
    {
      #pragma unroll
      for (int mt = 0; mt < 2; ++mt) {
        v16h A[8];
        #pragma unroll
        for (int ks = 0; ks < 8; ++ks) {
          const _Float16* p = &h1[(mt * 16 + lrow) * D1_ + ks * 32 + lhi * 16];
          v8h lo = *(const v8h*)p;
          v8h hi = *(const v8h*)(p + 8);
          #pragma unroll
          for (int e = 0; e < 8; ++e) { A[ks][e] = lo[e]; A[ks][e + 8] = hi[e]; }
        }
        #pragma unroll
        for (int ntl = 0; ntl < 4; ++ntl) {
          int nt = wave * 4 + ntl;
          int ncol = nt * 16 + lrow;
          const _Float16* wp = w2t + (size_t)ncol * D1_ + lhi * 16;
          v8f acc = {};
          #pragma unroll
          for (int ks = 0; ks < 8; ++ks) {
            v16h Bv = *(const v16h*)(wp + ks * 32);
            acc = __builtin_amdgcn_wmma_f32_16x16x32_f16(false, A[ks], false, Bv,
                                                         (short)0, acc, false, false);
          }
          float bias = b2[ncol];
          #pragma unroll
          for (int r8 = 0; r8 < 8; ++r8) {
            int row = mt * 16 + lhi * 8 + r8;
            h2[row * D2_ + ncol] = (_Float16)(acc[r8] + bias);
          }
        }
      }
    }
    __syncthreads();

    // ---- layer 3: (32x512) @ (512x64), v_wmma_f32_16x16x32_f16 ----
    {
      int mt = wave >> 2, nt = wave & 3;    // 8 tiles == 8 waves
      int ncol = nt * 16 + lrow;
      const _Float16* wp = w3t + (size_t)ncol * D2_ + lhi * 16;
      v8f acc = {};
      #pragma unroll
      for (int ks = 0; ks < 16; ++ks) {
        const _Float16* p = &h2[(mt * 16 + lrow) * D2_ + ks * 32 + lhi * 16];
        v8h lo = *(const v8h*)p;
        v8h hi = *(const v8h*)(p + 8);
        v16h Av;
        #pragma unroll
        for (int e = 0; e < 8; ++e) { Av[e] = lo[e]; Av[e + 8] = hi[e]; }
        v16h Bv = *(const v16h*)(wp + ks * 32);
        acc = __builtin_amdgcn_wmma_f32_16x16x32_f16(false, Av, false, Bv,
                                                     (short)0, acc, false, false);
      }
      float bias = b3[ncol];
      #pragma unroll
      for (int r8 = 0; r8 < 8; ++r8) {
        int row = mt * 16 + lhi * 8 + r8;
        tl[row * 64 + ncol] = acc[r8] + bias;
      }
    }
    __syncthreads();

    // ---- normalize quats + votes = capsule (x) t -> LDS ----
    for (int q = tid; q < CH * M_; q += 256) {
      int row = q >> 4, m = q & 15;
      float t0 = tl[row * 64 + m * 4 + 0];
      float t1 = tl[row * 64 + m * 4 + 1];
      float t2 = tl[row * 64 + m * 4 + 2];
      float t3 = tl[row * 64 + m * 4 + 3];
      float nrm = sqrtf(t0*t0 + t1*t1 + t2*t2 + t3*t3);
      float inv = 1.0f / fmaxf(nrm, 1e-12f);
      t0 *= inv; t1 *= inv; t2 *= inv; t3 *= inv;
      int fr = chunk * CH + row;
      const float* cp = caps_in + ((size_t)bn * ROWS + fr) * 4;
      float a0 = cp[0], a1 = cp[1], a2 = cp[2], a3 = cp[3];
      float* vp = &votes_l[(m * ROWS + fr) * 4];
      vp[0] = a0*t0 - a1*t1 - a2*t2 - a3*t3;
      vp[1] = a0*t1 + t0*a1 + a2*t3 - a3*t2;
      vp[2] = a0*t2 + t0*a2 + a3*t1 - a1*t3;
      vp[3] = a0*t3 + t0*a3 + a1*t2 - a2*t1;
    }
    __syncthreads();
  }

  // ---- routing: wave w handles m = w and m = w+8; votes stay in LDS ----
  const float4* votes4 = (const float4*)votes_l;
  #pragma unroll
  for (int mm = 0; mm < 2; ++mm) {
    int m = wave + mm * 8;
    int g = bn * M_ + m;
    float v[8][4], aw[8];
    #pragma unroll
    for (int i = 0; i < 8; ++i) {
      int j = i * 32 + lane;
      float4 q = votes4[m * ROWS + j];
      v[i][0] = q.x; v[i][1] = q.y; v[i][2] = q.z; v[i][3] = q.w;
      aw[i] = a_l[j];
    }

    unsigned int base_seed = 0xC2B2AE35u ^ ((unsigned)g * 0x9E3779B9u);
    float caps[4];
    // initial quat_avg with weights = activation
    {
      float c[10];
      #pragma unroll
      for (int k = 0; k < 10; ++k) c[k] = 0.0f;
      #pragma unroll
      for (int i = 0; i < 8; ++i) cov_accum(c, aw[i], v[i]);
      cov_reduce32(c);
      power_iter(c, base_seed, caps);
    }
    // 3 rounds of sigmoid-distance re-assignment
    for (int it = 0; it < 3; ++it) {
      float c[10];
      #pragma unroll
      for (int k = 0; k < 10; ++k) c[k] = 0.0f;
      #pragma unroll
      for (int i = 0; i < 8; ++i) {
        float dist = quat_dist(caps, v[i]);
        float w = aw[i] / (1.0f + expf(dist));       // a * sigmoid(-dist)
        cov_accum(c, w, v[i]);
      }
      cov_reduce32(c);
      power_iter(c, base_seed + 0x85EBCA6Bu * (unsigned)(it + 1), caps);
    }
    // final activation = sigmoid(-sum(dist))
    float dsum = 0.0f;
    #pragma unroll
    for (int i = 0; i < 8; ++i) dsum += quat_dist(caps, v[i]);
    dsum += __shfl_xor(dsum, 1, 32);
    dsum += __shfl_xor(dsum, 2, 32);
    dsum += __shfl_xor(dsum, 4, 32);
    dsum += __shfl_xor(dsum, 8, 32);
    dsum += __shfl_xor(dsum, 16, 32);
    if (lane == 0) {
      float* cp = out + (size_t)g * 4;
      cp[0] = caps[0]; cp[1] = caps[1]; cp[2] = caps[2]; cp[3] = caps[3];
      out[(size_t)B_ * N_ * M_ * 4 + g] = 1.0f / (1.0f + expf(dsum));
    }
  }
}

// ---------------- launch ----------------

extern "C" void kernel_launch(void* const* d_in, const int* in_sizes, int n_in,
                              void* d_out, int out_size, void* d_ws, size_t ws_size,
                              hipStream_t stream) {
  (void)in_sizes; (void)n_in; (void)out_size; (void)ws_size;
  const float* input_x   = (const float*)d_in[0];
  const float* input_cap = (const float*)d_in[1];
  const float* input_a   = (const float*)d_in[2];
  const float* W1 = (const float*)d_in[3];
  const float* b1 = (const float*)d_in[4];
  const float* W2 = (const float*)d_in[5];
  const float* b2 = (const float*)d_in[6];
  const float* W3 = (const float*)d_in[7];
  const float* b3 = (const float*)d_in[8];
  float* out = (float*)d_out;

  // workspace layout (all naturally aligned; hipMalloc base is 256B aligned)
  float* miu = (float*)d_ws;                               // 4*256*8*4      = 32768 f32
  float* x3  = miu + (size_t)B_ * N_ * NC_ * 4;            // 4*256*32*8*3   = 786432 f32
  _Float16* w2t = (_Float16*)(x3 + (size_t)B_ * N_ * K_ * NC_ * 3);  // 512*256 f16
  _Float16* w3t = w2t + (size_t)D2_ * D1_;                 // 64*512 f16

  qenet_prep_weights<<<(D2_ * D1_ + 64 * D2_ + 255) / 256, 256, 0, stream>>>(W2, W3, w2t, w3t);
  qenet_miu<<<(B_ * N_ * NC_ + 255) / 256, 256, 0, stream>>>(input_cap, miu);
  qenet_rotate<<<(B_ * N_ * K_ * NC_ + 255) / 256, 256, 0, stream>>>(input_x, miu, x3);
  qenet_mlp_route<<<B_ * N_, 256, 0, stream>>>(x3, W1, b1, w2t, b2, w3t, b3,
                                               input_cap, input_a, out);
}
